// LENet_79216376808037
// MI455X (gfx1250) — compile-verified, matrix operand order checked by reading, not run on hardware
//
#include <hip/hip_runtime.h>
#include <hip/hip_bf16.h>
#include <math.h>

typedef __attribute__((ext_vector_type(16))) _Float16 v16h;
typedef __attribute__((ext_vector_type(8)))  float    v8f;

#define NGRAPH 128

// K offset inside a 32-wide K chunk for f16 A/B WMMA fragments.
// ISA 7.12.2 (16-bit A 16x32): VGPR r, lane-half h:
//   r<4 : K = 2r + 8h (+bit0), r>=4 : K = 16 + 2(r-4) + 8h (+bit0)
__device__ __forceinline__ int wmma_koff(int j, int half) {
    int r = j >> 1;
    int base = (r < 4) ? (2 * r) : (16 + 2 * (r - 4));
    return base + 8 * half + (j & 1);
}

// ---------------------------------------------------------------------------
// Kernel 0: pack [w1|w2|w3] (fp32, row-major [DIN, DOUT]) into f16 WMMA
// B-fragment order: frag[(nt*NK + kk)*32 + lane][j]. Runs once per launch;
// weights are L2-resident and reused by all 3125 M-tile waves.
// ---------------------------------------------------------------------------
template <int DIN, int DOUT>
__global__ __launch_bounds__(256) void pack_weights_kernel(
    const float* __restrict__ w1, const float* __restrict__ w2,
    const float* __restrict__ w3, _Float16* __restrict__ out) {
    constexpr int NK    = DIN / 32;
    constexpr int NTILE = (3 * DOUT) / 16;
    constexpr int TPD   = DOUT / 16;
    constexpr int TOTAL = NTILE * NK * 32 * 16;
    int idx = blockIdx.x * blockDim.x + threadIdx.x;
    if (idx >= TOTAL) return;
    int j    = idx & 15;
    int lane = (idx >> 4) & 31;
    int kk   = (idx >> 9) % NK;
    int nt   = (idx >> 9) / NK;
    int half = lane >> 4, colIn = lane & 15;
    int sel = nt / TPD, nbase = (nt % TPD) * 16;
    const float* w = (sel == 0) ? w1 : (sel == 1 ? w2 : w3);
    out[idx] = (_Float16)w[(kk * 32 + wmma_koff(j, half)) * DOUT + nbase + colIn];
}

// ---------------------------------------------------------------------------
// Kernel 1: ew[e] = edge_attr[e,:16] . w_fc1 + b_fc1
// ---------------------------------------------------------------------------
__global__ __launch_bounds__(256) void edge_weight_kernel(
    const float* __restrict__ ea, const float* __restrict__ wfc,
    const float* __restrict__ bfc, float* __restrict__ ew, int E) {
    int e = blockIdx.x * blockDim.x + threadIdx.x;
    if (e >= E) return;
    const float* p = ea + e * 16;
    float s = bfc[0];
#pragma unroll
    for (int k = 0; k < 16; ++k) s += p[k] * wfc[k];
    ew[e] = s;
}

// ---------------------------------------------------------------------------
// Kernel 2: fused per-layer GEMM via f16 WMMA, f32 accumulate.
//   A   = elu?(x) @ w1 + b1
//   B   = elu?(x) @ w2
//   AGG = elu?(x) @ w3 + b3    (scatter kernel accumulates on top)
// ELU of the previous layer's agg is fused into the A-fragment conversion.
// One wave per 16-node tile; A fragments cached in VGPRs across all col tiles.
// ---------------------------------------------------------------------------
template <int DIN, int DOUT, bool ELU_IN>
__global__ __launch_bounds__(256) void leconv_gemm_kernel(
    const float* __restrict__ x, const _Float16* __restrict__ wpack,
    const float* __restrict__ b1, const float* __restrict__ b3,
    float* __restrict__ A, float* __restrict__ B, float* __restrict__ AGG,
    int mtiles) {
    constexpr int NK    = DIN / 32;        // K chunks of 32
    constexpr int NTILE = (3 * DOUT) / 16; // 16-wide output column tiles
    constexpr int TPD   = DOUT / 16;       // tiles per weight matrix

    int lane = threadIdx.x & 31;
    int wave = blockIdx.x * (blockDim.x >> 5) + (threadIdx.x >> 5);
    if (wave >= mtiles) return;            // wave-uniform: EXEC all-1 for WMMA

    int m0    = wave * 16;
    int half  = lane >> 4;
    int colIn = lane & 15;

    // Cache this wave's 16xDIN A slab as f16 fragments (fused ELU on input)
    v16h acache[NK];
    {
        const float* xr = x + (long long)(m0 + (lane & 15)) * DIN;
#pragma unroll
        for (int kk = 0; kk < NK; ++kk) {
#pragma unroll
            for (int j = 0; j < 16; ++j) {
                float v = xr[kk * 32 + wmma_koff(j, half)];
                if (ELU_IN) v = (v > 0.f) ? v : expm1f(v);
                acache[kk][j] = (_Float16)v;
            }
        }
    }

    const v16h* wp = reinterpret_cast<const v16h*>(wpack);
#pragma unroll
    for (int nt = 0; nt < NTILE; ++nt) {
        const int sel   = nt / TPD;           // 0:w1 1:w2 2:w3 (folds at -O3)
        const int nbase = (nt % TPD) * 16;

        v8f c = {0.f, 0.f, 0.f, 0.f, 0.f, 0.f, 0.f, 0.f};
#pragma unroll
        for (int kk = 0; kk < NK; ++kk) {
            v16h bfrag = wp[(nt * NK + kk) * 32 + lane];  // 32B packed load
            c = __builtin_amdgcn_wmma_f32_16x16x32_f16(
                    false, acache[kk], false, bfrag, (short)0, c, false, false);
        }

        float bias = 0.f;
        if (sel == 0) bias = b1[nbase + colIn];
        if (sel == 2) bias = b3[nbase + colIn];
        float* outp = (sel == 0) ? A : (sel == 1 ? B : AGG);
        // D layout: VGPR r -> M = r (lanes 0-15) / r+8 (lanes 16-31), N = lane%16
#pragma unroll
        for (int r = 0; r < 8; ++r) {
            int m = m0 + r + (half ? 8 : 0);
            outp[(long long)m * DOUT + nbase + colIn] = c[r] + bias;
        }
    }
}

// ---------------------------------------------------------------------------
// Kernel 3: edge scatter  AGG[dst] += ew * (A[src] - B[dst]), float4 per thread
// ---------------------------------------------------------------------------
template <int DOUT>
__global__ __launch_bounds__(256) void edge_scatter_kernel(
    const int* __restrict__ ei, const float* __restrict__ ew,
    const float* __restrict__ A, const float* __restrict__ B,
    float* __restrict__ AGG, int E) {
    constexpr int FQ = DOUT / 4;           // float4 groups per edge
    int t = blockIdx.x * blockDim.x + threadIdx.x;
    if (t >= E * FQ) return;
    int e = t / FQ;
    int f = (t % FQ) * 4;
    int src = ei[e];
    int dst = ei[E + e];
    float w = ew[e];
    float4 a = *(const float4*)(A + src * DOUT + f);
    float4 b = *(const float4*)(B + dst * DOUT + f);
    float* g = AGG + dst * DOUT + f;
    atomicAdd(g + 0, w * (a.x - b.x));
    atomicAdd(g + 1, w * (a.y - b.y));
    atomicAdd(g + 2, w * (a.z - b.z));
    atomicAdd(g + 3, w * (a.w - b.w));
}

// ---------------------------------------------------------------------------
// Kernels 4-6: final ELU + segment-max pool (order-preserving uint encoding)
// ---------------------------------------------------------------------------
__device__ __forceinline__ unsigned enc_f32(float f) {
    unsigned u = __float_as_uint(f);
    return (u & 0x80000000u) ? ~u : (u | 0x80000000u);
}
__device__ __forceinline__ float dec_f32(unsigned u) {
    return (u & 0x80000000u) ? __uint_as_float(u ^ 0x80000000u)
                             : __uint_as_float(~u);
}

__global__ __launch_bounds__(256) void maxbuf_init_kernel(unsigned* __restrict__ m, int n) {
    int i = blockIdx.x * blockDim.x + threadIdx.x;
    if (i < n) m[i] = 0x007FFFFFu;  // enc(-inf)
}

__global__ __launch_bounds__(256) void elu_maxpool_kernel(
    const float* __restrict__ agg, const int* __restrict__ batch,
    unsigned* __restrict__ maxbuf, int n) {  // n = N*64
    int t = blockIdx.x * blockDim.x + threadIdx.x;
    if (t >= n) return;
    int i = t >> 6, f = t & 63;
    float v = agg[t];
    v = (v > 0.f) ? v : expm1f(v);
    atomicMax(&maxbuf[batch[i] * 64 + f], enc_f32(v));
}

__global__ __launch_bounds__(256) void max_decode_kernel(
    const unsigned* __restrict__ maxbuf, float* __restrict__ out, int n) {
    int i = blockIdx.x * blockDim.x + threadIdx.x;
    if (i < n) out[i] = dec_f32(maxbuf[i]);
}

// ---------------------------------------------------------------------------
extern "C" void kernel_launch(void* const* d_in, const int* in_sizes, int n_in,
                              void* d_out, int out_size, void* d_ws, size_t ws_size,
                              hipStream_t stream) {
    (void)n_in; (void)ws_size; (void)out_size;
    const float* x_in  = (const float*)d_in[0];
    const int*   ei    = (const int*)d_in[1];
    const float* eattr = (const float*)d_in[2];
    const int*   batch = (const int*)d_in[3];
    const float* w_fc1 = (const float*)d_in[4];
    const float* b_fc1 = (const float*)d_in[5];
    // per-layer params at 6 + 5*i : w1, b1, w2, w3, b3
    const float* w1[3]; const float* b1[3]; const float* w2[3];
    const float* w3[3]; const float* b3[3];
    for (int i = 0; i < 3; ++i) {
        w1[i] = (const float*)d_in[6 + 5 * i + 0];
        b1[i] = (const float*)d_in[6 + 5 * i + 1];
        w2[i] = (const float*)d_in[6 + 5 * i + 2];
        w3[i] = (const float*)d_in[6 + 5 * i + 3];
        b3[i] = (const float*)d_in[6 + 5 * i + 4];
    }
    const int N = in_sizes[0] / 64;     // 50000 (divisible by 16)
    const int E = in_sizes[1] / 2;      // 800000
    const int mtiles = N / 16;

    // Workspace layout
    float* ws    = (float*)d_ws;
    float* ew    = ws;                       // E floats
    float* Abuf  = ew + E;                   // N*128
    float* Bbuf  = Abuf + (size_t)N * 128;   // N*128
    float* AGGa  = Bbuf + (size_t)N * 128;   // N*128
    float* AGGb  = AGGa + (size_t)N * 128;   // N*128
    _Float16* wpk0 = (_Float16*)(AGGb + (size_t)N * 128);  //  64*192 halfs
    _Float16* wpk1 = wpk0 + 64 * 192;                      //  64*384 halfs
    _Float16* wpk2 = wpk1 + 64 * 384;                      // 128*192 halfs
    unsigned* maxbuf = (unsigned*)(wpk2 + 128 * 192);      // NGRAPH*64

    const int TPB = 256;
    const int gemm_blocks = (mtiles + 7) / 8;   // 8 waves per block

    // prep: edge weights + packed f16 weight fragments
    edge_weight_kernel<<<(E + TPB - 1) / TPB, TPB, 0, stream>>>(eattr, w_fc1, b_fc1, ew, E);
    pack_weights_kernel<64, 64><<<(64 * 192 + TPB - 1) / TPB, TPB, 0, stream>>>(
        w1[0], w2[0], w3[0], wpk0);
    pack_weights_kernel<64, 128><<<(64 * 384 + TPB - 1) / TPB, TPB, 0, stream>>>(
        w1[1], w2[1], w3[1], wpk1);
    pack_weights_kernel<128, 64><<<(128 * 192 + TPB - 1) / TPB, TPB, 0, stream>>>(
        w1[2], w2[2], w3[2], wpk2);

    // ---- Layer 0: 64 -> 64 (input: raw x) ----
    leconv_gemm_kernel<64, 64, false><<<gemm_blocks, TPB, 0, stream>>>(
        x_in, wpk0, b1[0], b3[0], Abuf, Bbuf, AGGa, mtiles);
    edge_scatter_kernel<64><<<(E * 16 + TPB - 1) / TPB, TPB, 0, stream>>>(
        ei, ew, Abuf, Bbuf, AGGa, E);

    // ---- Layer 1: 64 -> 128 (input: elu(AGGa), fused) ----
    leconv_gemm_kernel<64, 128, true><<<gemm_blocks, TPB, 0, stream>>>(
        AGGa, wpk1, b1[1], b3[1], Abuf, Bbuf, AGGb, mtiles);
    edge_scatter_kernel<128><<<(E * 32 + TPB - 1) / TPB, TPB, 0, stream>>>(
        ei, ew, Abuf, Bbuf, AGGb, E);

    // ---- Layer 2: 128 -> 64 (input: elu(AGGb), fused) ----
    leconv_gemm_kernel<128, 64, true><<<gemm_blocks, TPB, 0, stream>>>(
        AGGb, wpk2, b1[2], b3[2], Abuf, Bbuf, AGGa, mtiles);
    edge_scatter_kernel<64><<<(E * 16 + TPB - 1) / TPB, TPB, 0, stream>>>(
        ei, ew, Abuf, Bbuf, AGGa, E);

    // ---- final ELU + global max pool ----
    const int pooln = NGRAPH * 64;
    maxbuf_init_kernel<<<(pooln + TPB - 1) / TPB, TPB, 0, stream>>>(maxbuf, pooln);
    elu_maxpool_kernel<<<(N * 64 + TPB - 1) / TPB, TPB, 0, stream>>>(AGGa, batch, maxbuf, N * 64);
    max_decode_kernel<<<(pooln + TPB - 1) / TPB, TPB, 0, stream>>>(maxbuf, (float*)d_out, pooln);
}
